// Multi_CrossAttention_63763084476524
// MI455X (gfx1250) — compile-verified, matrix operand order checked by reading, not run on hardware
//
#include <hip/hip_runtime.h>
#include <hip/hip_bf16.h>

// ---------------------------------------------------------------------------
// Multi-head cross-attention for MI455X (gfx1250).
// wave32 + v_wmma_f32_16x16x32_bf16 everywhere; global->LDS tile staging via
// the Tensor Data Mover (tensor_load_to_lds + s_wait_tensorcnt) when the
// builtin is available, cooperative b128 copies otherwise.
// ---------------------------------------------------------------------------

typedef __bf16 bf16_t;
typedef __attribute__((ext_vector_type(16))) __bf16 v16bf;
typedef __attribute__((ext_vector_type(8)))  __bf16 v8bf;
typedef __attribute__((ext_vector_type(8)))  float  v8f;
typedef unsigned int v4u  __attribute__((ext_vector_type(4)));
typedef int          v8i_t __attribute__((ext_vector_type(8)));
typedef int          v4i_t __attribute__((ext_vector_type(4)));

#define HEADS 16
#define DH    64
#define DMODEL 1024

#if defined(__has_builtin)
#  if __has_builtin(__builtin_amdgcn_tensor_load_to_lds)
#    define HAVE_TDM 1
#  else
#    define HAVE_TDM 0
#  endif
#else
#  define HAVE_TDM 0
#endif

// ---- fragment helpers -----------------------------------------------------
// A (16xK) / B (Kx16) bf16 fragment, per ISA 16-bit layout:
//   lane < 16 : m/n = lane,    k = kbase + {0..7} and {16..23}
//   lane >= 16: m/n = lane-16, k = kbase + {8..15} and {24..31}
__device__ __forceinline__ const bf16_t* frag_addr(const bf16_t* base, int stride,
                                                   int mn_base, int k_base) {
    int lane = threadIdx.x & 31;
    int mn = mn_base + (lane & 15);
    int k0 = k_base + ((lane >> 4) << 3);
    return base + (size_t)mn * stride + k0;
}

__device__ __forceinline__ v16bf load_frag16(const bf16_t* p) {
    v8bf lo = *(const v8bf*)p;          // k0 .. k0+7
    v8bf hi = *(const v8bf*)(p + 16);   // k0+16 .. k0+23
    v16bf r;
#pragma unroll
    for (int i = 0; i < 8; ++i) { r[i] = lo[i]; r[i + 8] = hi[i]; }
    return r;
}

__device__ __forceinline__ v8f wmma_bf16(v16bf a, v16bf b, v8f c) {
    return __builtin_amdgcn_wmma_f32_16x16x32_bf16(
        /*neg_a=*/false, a, /*neg_b=*/false, b,
        /*c_mod=*/(short)0, c, /*reuse_a=*/false, /*reuse_b=*/false);
}

// ---- TDM 2D tile load: global (strided rows) -> LDS (packed rows) ----------
#if HAVE_TDM
__device__ __forceinline__ void tdm_load_2d(unsigned lds_addr, const bf16_t* g,
                                            unsigned t_row_len, unsigned t_nrows,
                                            unsigned tile_w, unsigned tile_h,
                                            unsigned long long row_stride_elems) {
    unsigned long long ga = (unsigned long long)(size_t)g;
    v4u g0;
    g0.x = 1u;                                          // count=1, no gather
    g0.y = lds_addr;                                    // lds_addr [63:32]
    g0.z = (unsigned)ga;                                // global_addr [95:64]
    g0.w = (unsigned)((ga >> 32) & 0x1FFFFFFu) | (2u << 30);  // addr hi | type=2
    v8i_t g1;
    g1[0] = (int)(1u << 16);                            // data_size = 2 bytes
    g1[1] = (int)((t_row_len & 0xFFFFu) << 16);         // tensor_dim0 lo
    g1[2] = (int)((t_row_len >> 16) | ((t_nrows & 0xFFFFu) << 16));
    g1[3] = (int)((t_nrows >> 16) | (tile_w << 16));    // tile_dim0
    g1[4] = (int)tile_h;                                // tile_dim1 (tile_dim2=0)
    g1[5] = (int)(unsigned)(row_stride_elems & 0xFFFFFFFFull);   // dim0 stride lo
    g1[6] = (int)(unsigned)(row_stride_elems >> 32);             // stride hi
    g1[7] = 0;
    v4i_t z4 = {0, 0, 0, 0};
    v8i_t z8 = {0, 0, 0, 0, 0, 0, 0, 0};
    // 6-arg toolchain variant: (g0, g1, g2, g3, extra, cpol)
    __builtin_amdgcn_tensor_load_to_lds(g0, g1, z4, z4, z8, 0);
}
#endif

// Stage a [rows x row_elems] bf16 tile (row stride g_stride elems) into LDS.
// Must be bracketed by __syncthreads() in the caller.
__device__ __forceinline__ void stage_tile(bf16_t* lds, const bf16_t* g,
                                           int row_elems, int rows,
                                           int g_stride, unsigned tensor_rows) {
#if HAVE_TDM
    int wid = __builtin_amdgcn_readfirstlane((int)(threadIdx.x >> 5));
    if (wid == 0) {
        tdm_load_2d((unsigned)(size_t)lds, g,
                    (unsigned)g_stride, tensor_rows,
                    (unsigned)row_elems, (unsigned)rows,
                    (unsigned long long)g_stride);
    }
#else
    int chunks_per_row = row_elems >> 3;
    int nchunks = rows * chunks_per_row;
    for (int i = threadIdx.x; i < nchunks; i += blockDim.x) {
        int r = i / chunks_per_row, c = i - r * chunks_per_row;
        ((v8bf*)lds)[i] = *(const v8bf*)(g + (size_t)r * g_stride + c * 8);
    }
#endif
}

__device__ __forceinline__ void stage_wait() {
#if HAVE_TDM
    __builtin_amdgcn_s_wait_tensorcnt(0);   // no-op for non-issuing waves
#endif
}

// ---- f32 -> bf16 cast -----------------------------------------------------
__global__ void cast_bf16_kernel(const float* __restrict__ src,
                                 bf16_t* __restrict__ dst, int n) {
    int i = blockIdx.x * blockDim.x + threadIdx.x;
    int stride = gridDim.x * blockDim.x;
    for (; i < n; i += stride) dst[i] = (bf16_t)src[i];
}

// ---- GEMM: C[M,N] = A[M,K] @ W[N,K]^T -------------------------------------
// 128 threads = 4 waves; wave covers 16(M) x 64(N); block covers 64 x 64.
// Per k-step the block TDM-stages a 64x32 A tile and 64x32 W tile into LDS.
// mode 0: store bf16 head-split  [B,H,S,DH]      (q, k)
// mode 1: store bf16 transposed  [B,H,DH,S]      (vT)
// mode 2: store f32 + bias       [M,N]           (final output)
__global__ void gemm_bf16_kernel(const bf16_t* __restrict__ A,
                                 const bf16_t* __restrict__ W,
                                 void* __restrict__ out,
                                 const float* __restrict__ bias,
                                 int M, int N, int K, int mode, int S) {
    __shared__ __align__(16) bf16_t ldsA[64 * 32];
    __shared__ __align__(16) bf16_t ldsW[64 * 32];

    int lane = threadIdx.x & 31;
    int wave = threadIdx.x >> 5;
    int mblock = blockIdx.x * 64;
    int ntile  = blockIdx.y * 64;
    int mtile  = mblock + wave * 16;

    v8f acc[4];
#pragma unroll
    for (int j = 0; j < 4; ++j) acc[j] = (v8f)(0.0f);

    for (int kb = 0; kb < K; kb += 32) {
        __syncthreads();                                   // previous tile consumed
        stage_tile(ldsA, A + (size_t)mblock * K + kb, 32, 64, K, (unsigned)M);
        stage_tile(ldsW, W + (size_t)ntile  * K + kb, 32, 64, K, (unsigned)N);
        stage_wait();
        __syncthreads();                                   // tile ready

        v16bf a = load_frag16(frag_addr(ldsA, 32, wave * 16, 0));
#pragma unroll
        for (int j = 0; j < 4; ++j) {
            v16bf b = load_frag16(frag_addr(ldsW, 32, j * 16, 0));
            acc[j] = wmma_bf16(a, b, acc[j]);
        }
    }

    int half = lane >> 4, col = lane & 15;
#pragma unroll
    for (int j = 0; j < 4; ++j) {
#pragma unroll
        for (int r = 0; r < 8; ++r) {
            int m = mtile + half * 8 + r;
            int n = ntile + j * 16 + col;
            float v = acc[j][r];
            if (mode == 2) {
                ((float*)out)[(size_t)m * N + n] = v + bias[n];
            } else {
                int h = n >> 6, d = n & 63;         // head, dim within head
                int bb = m / S, s = m % S;
                if (mode == 0) {
                    ((bf16_t*)out)[(((size_t)bb * HEADS + h) * S + s) * DH + d] = (bf16_t)v;
                } else {  // transposed V
                    ((bf16_t*)out)[(((size_t)bb * HEADS + h) * DH + d) * S + s] = (bf16_t)v;
                }
            }
        }
    }
}

// ---- flash-style attention ------------------------------------------------
// q: [B,H,Sq,DH] bf16, k: [B,H,Skv,DH] bf16, vT: [B,H,DH,Skv] bf16
// mask: [B,1,Sq,Skv] int32; ctx out: [B*Sq, DMODEL] bf16
// 128 threads = 4 waves on one (b,h); each wave owns a 16-row q tile.
// Per kv-step the block TDM-stages the 32x64 K tile and 64x32 vT tile.
__global__ void attn_kernel(const bf16_t* __restrict__ q,
                            const bf16_t* __restrict__ k,
                            const bf16_t* __restrict__ vT,
                            const int* __restrict__ mask,
                            bf16_t* __restrict__ ctx,
                            int Sq, int Skv) {
    __shared__ __align__(16) bf16_t ldsK[32 * 64];     // [kv 32][d 64]
    __shared__ __align__(16) bf16_t ldsV[64 * 32];     // [d 64][kv 32]
    __shared__ __align__(16) bf16_t Plds[4][16 * 32];  // per-wave P tile

    int lane = threadIdx.x & 31;
    int wave = threadIdx.x >> 5;
    int half = lane >> 4, col = lane & 15;

    int qblocks = Sq / 64;
    int bh = blockIdx.x / qblocks;           // b*H + h
    int q64 = blockIdx.x % qblocks;
    int bb = bh >> 4;
    int h  = bh & 15;
    int qbase = q64 * 64 + wave * 16;

    const bf16_t* qp = q  + (size_t)bh * Sq  * DH;
    const bf16_t* kp = k  + (size_t)bh * Skv * DH;
    const bf16_t* vp = vT + (size_t)bh * DH  * Skv;

    // Q tile stays in registers: 16 x 64 = two K-chunks of 32
    v16bf qa0 = load_frag16(frag_addr(qp, DH, qbase, 0));
    v16bf qa1 = load_frag16(frag_addr(qp, DH, qbase, 32));

    v8f o0 = (v8f)(0.0f), o1 = (v8f)(0.0f), o2 = (v8f)(0.0f), o3 = (v8f)(0.0f);
    float mrow[8], lrow[8];
#pragma unroll
    for (int r = 0; r < 8; ++r) { mrow[r] = -3.0e38f; lrow[r] = 0.0f; }

    const float scale = 0.125f;  // 1/sqrt(64)
    bf16_t* pw = &Plds[wave][0];

    for (int kb = 0; kb < Skv; kb += 32) {
        __syncthreads();
        // K tile rows are contiguous [32][64]; vT tile is 64 rows of 32, stride Skv
        stage_tile(ldsK, kp + (size_t)kb * DH, 64, 32, DH, (unsigned)Skv);
        stage_tile(ldsV, vp + kb, 32, 64, Skv, (unsigned)DH);
        stage_wait();
        __syncthreads();

        // ---- scores S[16 q, 32 kv] = Q @ K^T --------------------------------
        v8f s0 = (v8f)(0.0f), s1 = (v8f)(0.0f);
        {
            v16bf b0 = load_frag16(frag_addr(ldsK, DH, 0,  0));
            v16bf b1 = load_frag16(frag_addr(ldsK, DH, 16, 0));
            s0 = wmma_bf16(qa0, b0, s0);
            s1 = wmma_bf16(qa0, b1, s1);
            b0 = load_frag16(frag_addr(ldsK, DH, 0,  32));
            b1 = load_frag16(frag_addr(ldsK, DH, 16, 32));
            s0 = wmma_bf16(qa1, b0, s0);
            s1 = wmma_bf16(qa1, b1, s1);
        }

        // ---- online softmax (mask -> -1e9 BEFORE scale; faithful quirk) -----
#pragma unroll
        for (int r = 0; r < 8; ++r) {
            int row = qbase + half * 8 + r;
            const int* mp = mask + ((size_t)bb * Sq + row) * Skv + kb;
            float x0 = (mp[col]      == 0) ? -1.0e9f : s0[r];
            float x1 = (mp[col + 16] == 0) ? -1.0e9f : s1[r];
            x0 *= scale; x1 *= scale;

            float mx = fmaxf(x0, x1);
            mx = fmaxf(mx, __shfl_xor(mx, 1, 32));
            mx = fmaxf(mx, __shfl_xor(mx, 2, 32));
            mx = fmaxf(mx, __shfl_xor(mx, 4, 32));
            mx = fmaxf(mx, __shfl_xor(mx, 8, 32));

            float mnew  = fmaxf(mrow[r], mx);
            float alpha = __expf(mrow[r] - mnew);
            float p0 = __expf(x0 - mnew);
            float p1 = __expf(x1 - mnew);

            float ps = p0 + p1;
            ps += __shfl_xor(ps, 1, 32);
            ps += __shfl_xor(ps, 2, 32);
            ps += __shfl_xor(ps, 4, 32);
            ps += __shfl_xor(ps, 8, 32);

            lrow[r] = lrow[r] * alpha + ps;
            mrow[r] = mnew;
            o0[r] *= alpha; o1[r] *= alpha; o2[r] *= alpha; o3[r] *= alpha;

            // stage P (D-layout -> LDS row-major 16x32) for re-load as A-frag
            pw[(half * 8 + r) * 32 + col]      = (bf16_t)p0;
            pw[(half * 8 + r) * 32 + col + 16] = (bf16_t)p1;
        }

        // ---- O += P @ V  (V tile in LDS, B-fragments contiguous) ------------
        v16bf pa = load_frag16(frag_addr(pw, 32, 0, 0));
        o0 = wmma_bf16(pa, load_frag16(frag_addr(ldsV, 32, 0,  0)), o0);
        o1 = wmma_bf16(pa, load_frag16(frag_addr(ldsV, 32, 16, 0)), o1);
        o2 = wmma_bf16(pa, load_frag16(frag_addr(ldsV, 32, 32, 0)), o2);
        o3 = wmma_bf16(pa, load_frag16(frag_addr(ldsV, 32, 48, 0)), o3);
    }

    // ---- epilogue: normalize, merge heads back to [B*Sq, DMODEL] -----------
#pragma unroll
    for (int r = 0; r < 8; ++r) {
        int row = qbase + half * 8 + r;
        float inv = 1.0f / lrow[r];
        size_t base = ((size_t)bb * Sq + row) * DMODEL + h * DH + col;
        ctx[base + 0 * 16] = (bf16_t)(o0[r] * inv);
        ctx[base + 1 * 16] = (bf16_t)(o1[r] * inv);
        ctx[base + 2 * 16] = (bf16_t)(o2[r] * inv);
        ctx[base + 3 * 16] = (bf16_t)(o3[r] * inv);
    }
}

// ---------------------------------------------------------------------------
extern "C" void kernel_launch(void* const* d_in, const int* in_sizes, int n_in,
                              void* d_out, int out_size, void* d_ws, size_t ws_size,
                              hipStream_t stream) {
    const float* x    = (const float*)d_in[0];
    const float* y    = (const float*)d_in[1];
    const int*   mask = (const int*)  d_in[2];
    const float* Wq   = (const float*)d_in[3];
    const float* Wk   = (const float*)d_in[4];
    const float* Wv   = (const float*)d_in[5];
    const float* Wo   = (const float*)d_in[6];
    const float* bo   = (const float*)d_in[7];
    float* out = (float*)d_out;

    const int B   = 2;
    const int Sq  = in_sizes[0] / (DMODEL * B);
    const int Skv = in_sizes[1] / (DMODEL * B);
    const int Mq  = B * Sq;     // 4096
    const int Mkv = B * Skv;    // 4096

    // ---- workspace carve (bf16 buffers), 256B aligned chunks ---------------
    char* ws = (char*)d_ws;
    size_t off = 0;
    auto carve = [&](size_t bytes) {
        void* p = ws + off;
        off += (bytes + 255) & ~(size_t)255;
        return p;
    };
    bf16_t* xb  = (bf16_t*)carve((size_t)Mq  * DMODEL * 2);
    bf16_t* yb  = (bf16_t*)carve((size_t)Mkv * DMODEL * 2);
    bf16_t* Wqb = (bf16_t*)carve((size_t)DMODEL * DMODEL * 2);
    bf16_t* Wkb = (bf16_t*)carve((size_t)DMODEL * DMODEL * 2);
    bf16_t* Wvb = (bf16_t*)carve((size_t)DMODEL * DMODEL * 2);
    bf16_t* Wob = (bf16_t*)carve((size_t)DMODEL * DMODEL * 2);
    bf16_t* qb  = (bf16_t*)carve((size_t)Mq  * DMODEL * 2);  // [B,H,Sq,DH]
    bf16_t* kb_ = (bf16_t*)carve((size_t)Mkv * DMODEL * 2);  // [B,H,Skv,DH]
    bf16_t* vtb = (bf16_t*)carve((size_t)Mkv * DMODEL * 2);  // [B,H,DH,Skv]
    bf16_t* ctx = (bf16_t*)carve((size_t)Mq  * DMODEL * 2);  // [B*Sq,DMODEL]

    // ---- stage 1: casts -----------------------------------------------------
    auto cast = [&](const float* s, bf16_t* d, int n) {
        int blocks = (n + 1023) / 1024;
        cast_bf16_kernel<<<blocks, 256, 0, stream>>>(s, d, n);
    };
    cast(x,  xb,  Mq  * DMODEL);
    cast(y,  yb,  Mkv * DMODEL);
    cast(Wq, Wqb, DMODEL * DMODEL);
    cast(Wk, Wkb, DMODEL * DMODEL);
    cast(Wv, Wvb, DMODEL * DMODEL);
    cast(Wo, Wob, DMODEL * DMODEL);

    // ---- stage 2/3: projections --------------------------------------------
    dim3 blk(128);
    dim3 gq(Mq / 64, DMODEL / 64);
    dim3 gkv(Mkv / 64, DMODEL / 64);
    gemm_bf16_kernel<<<gq,  blk, 0, stream>>>(xb, Wqb, qb,  nullptr, Mq,  DMODEL, DMODEL, 0, Sq);
    gemm_bf16_kernel<<<gkv, blk, 0, stream>>>(yb, Wkb, kb_, nullptr, Mkv, DMODEL, DMODEL, 0, Skv);
    gemm_bf16_kernel<<<gkv, blk, 0, stream>>>(yb, Wvb, vtb, nullptr, Mkv, DMODEL, DMODEL, 1, Skv);

    // ---- stage 4: attention -------------------------------------------------
    int attn_blocks = B * HEADS * (Sq / 64);
    attn_kernel<<<attn_blocks, blk, 0, stream>>>(qb, kb_, vtb, mask, ctx, Sq, Skv);

    // ---- stage 5: output projection + bias ---------------------------------
    gemm_bf16_kernel<<<gq, blk, 0, stream>>>(ctx, Wob, out, bo, Mq, DMODEL, DMODEL, 2, Sq);
}
